// MultiHeadSelfAttention2d_6313601925672
// MI455X (gfx1250) — matrix operand order repaired and checked by reading, not compiled, not run on hardware
//
#include <hip/hip_runtime.h>

typedef unsigned short u16;
typedef __attribute__((ext_vector_type(16))) __bf16 v16bf;
typedef __attribute__((ext_vector_type(8)))  float  v8f;
typedef __attribute__((ext_vector_type(4)))  int    i32x4;

#define C_DIM 512
#define N_SP  1024
#define BATCH 16
#define HEADS 8
#define HDIM  64
#define SCALE_F 0.125f

#if __has_builtin(__builtin_amdgcn_global_load_async_to_lds_b128) && \
    __has_builtin(__builtin_amdgcn_s_wait_asynccnt)
#define USE_ASYNC 1
#else
#define USE_ASYNC 0
#endif

__device__ __forceinline__ u16 f2bf(float f) {
  union { float f; unsigned u; } c; c.f = f;
  unsigned u = c.u;
  u += 0x7fffu + ((u >> 16) & 1u);   // round to nearest even
  return (u16)(u >> 16);
}

// copy 8 bf16 (16 bytes) global -> LDS
__device__ __forceinline__ void cp128(u16* ldst, const u16* gsrc) {
#if USE_ASYNC
  __builtin_amdgcn_global_load_async_to_lds_b128(
      (__attribute__((address_space(1))) i32x4*)gsrc,
      (__attribute__((address_space(3))) i32x4*)ldst, 0, 0);
#else
  *(uint4*)ldst = *(const uint4*)gsrc;
#endif
}
__device__ __forceinline__ void async_wait() {
#if USE_ASYNC
  __builtin_amdgcn_s_wait_asynccnt(0);
#endif
}

__device__ __forceinline__ v8f bf16_wmma(const unsigned a[8], const unsigned b[8], v8f acc) {
  union { unsigned u[8]; v16bf v; } A, B;
#pragma unroll
  for (int i = 0; i < 8; ++i) { A.u[i] = a[i]; B.u[i] = b[i]; }
  return __builtin_amdgcn_wmma_f32_16x16x32_bf16(false, A.v, false, B.v, (short)0, acc, false, false);
}

__device__ __forceinline__ float half_max(float v) {
#pragma unroll
  for (int m = 1; m <= 8; m <<= 1) v = fmaxf(v, __shfl_xor(v, m, 32));
  return v;
}
__device__ __forceinline__ float half_sum(float v) {
#pragma unroll
  for (int m = 1; m <= 8; m <<= 1) v += __shfl_xor(v, m, 32);
  return v;
}

// ---------------------------------------------------------------- weight convert
__global__ __launch_bounds__(256) void cvt_bf16(const float* __restrict__ s,
                                                u16* __restrict__ d, int n) {
  for (int i = blockIdx.x * 256 + threadIdx.x; i < n; i += gridDim.x * 256)
    d[i] = f2bf(s[i]);
}

// ---------------------------------------------------------------- x transpose+convert
// x[b][c][n] fp32 -> xb[b][n][c] bf16  (32x32 LDS tile transpose)
__global__ __launch_bounds__(256) void cvt_x_t(const float* __restrict__ x,
                                               u16* __restrict__ xb) {
  __shared__ float t[32][33];
  const int b  = blockIdx.z;
  const int c0 = blockIdx.y * 32, n0 = blockIdx.x * 32;
  const int tl = threadIdx.x & 31, th = threadIdx.x >> 5;  // th in 0..7
#pragma unroll
  for (int i = 0; i < 4; ++i) {
    int c = th + i * 8;
    t[c][tl] = x[((size_t)b * C_DIM + c0 + c) * N_SP + n0 + tl];
  }
  __syncthreads();
#pragma unroll
  for (int i = 0; i < 4; ++i) {
    int n = th + i * 8;
    xb[((size_t)b * N_SP + n0 + n) * C_DIM + c0 + tl] = f2bf(t[tl][n]);
  }
}

// ---------------------------------------------------------------- QKV projection
// out[o,n] = sum_c W[o,c]*X[n,c] + bias[o]
// vmode=0 -> store [b][h][n][d] (Q,K); vmode=1 -> store [b][h][d][n] (V)
__global__ __launch_bounds__(256) void qkv_gemm(const u16* __restrict__ W,
                                                const float* __restrict__ bias,
                                                const u16* __restrict__ X,
                                                u16* __restrict__ out, int vmode) {
  __shared__ u16 lA[128 * 32];   // [m][k]  8 KB
  __shared__ u16 lB[128 * 32];   // [n][k]  8 KB  (k-pairs contiguous!)

  const int tid  = threadIdx.x;
  const int wave = tid >> 5, lane = tid & 31;
  const int mlo  = lane & 15, half = lane >> 4;
  const int b    = blockIdx.z;
  const int m0   = blockIdx.y * 128;
  const int n0   = blockIdx.x * 128;
  const int wm   = (wave >> 1) * 32;
  const int wn   = (wave & 1) * 64;

  const u16* Xb = X + (size_t)b * N_SP * C_DIM;
  v8f acc[2][4] = {};

  for (int k0 = 0; k0 < C_DIM; k0 += 32) {
    {   // stage A + B rows: direct row copies (async)
      int row = tid >> 1, off = (tid & 1) * 16;
      const u16* srcA = W + (size_t)(m0 + row) * C_DIM + k0 + off;
      const u16* srcB = Xb + (size_t)(n0 + row) * C_DIM + k0 + off;
      if (k0 + 32 < C_DIM) __builtin_prefetch(srcA + 32, 0, 1);
      cp128(lA + row * 32 + off,     srcA);
      cp128(lA + row * 32 + off + 8, srcA + 8);
      cp128(lB + row * 32 + off,     srcB);
      cp128(lB + row * 32 + off + 8, srcB + 8);
    }
    async_wait();
    __syncthreads();

    unsigned afr[2][8], bfr[4][8];
#pragma unroll
    for (int mi = 0; mi < 2; ++mi) {
      int m = wm + mi * 16 + mlo;
#pragma unroll
      for (int v = 0; v < 8; ++v) {
        int kk = (v >> 2) * 16 + half * 8 + (v & 3) * 2;
        afr[mi][v] = *(const unsigned*)(lA + m * 32 + kk);
      }
    }
#pragma unroll
    for (int ni = 0; ni < 4; ++ni) {
      int n = wn + ni * 16 + mlo;
#pragma unroll
      for (int v = 0; v < 8; ++v)
        bfr[ni][v] = *(const unsigned*)(lB + n * 32 + (v >> 2) * 16 + half * 8 + (v & 3) * 2);
    }
#pragma unroll
    for (int mi = 0; mi < 2; ++mi)
#pragma unroll
      for (int ni = 0; ni < 4; ++ni)
        acc[mi][ni] = bf16_wmma(afr[mi], bfr[ni], acc[mi][ni]);
    __syncthreads();
  }

#pragma unroll
  for (int mi = 0; mi < 2; ++mi) {
#pragma unroll
    for (int ni = 0; ni < 4; ++ni) {
      int n = n0 + wn + ni * 16 + mlo;
#pragma unroll
      for (int r = 0; r < 8; ++r) {
        int o = m0 + wm + mi * 16 + half * 8 + r;
        float val = acc[mi][ni][r] + bias[o];
        int h = o >> 6, d = o & 63;
        size_t idx = vmode
            ? (((size_t)b * HEADS + h) * HDIM + d) * N_SP + n   // [b][h][d][n]
            : (((size_t)b * HEADS + h) * N_SP + n) * HDIM + d;  // [b][h][n][d]
        out[idx] = f2bf(val);
      }
    }
  }
}

// ---------------------------------------------------------------- flash attention
// Q,K in [b][h][n][d]; V in [b][h][d][n]; writes O^T into [b][n][c]
__global__ __launch_bounds__(256) void flash_attn(const u16* __restrict__ Q,
                                                  const u16* __restrict__ K,
                                                  const u16* __restrict__ V,
                                                  u16* __restrict__ Ob) {
  __shared__ u16 lK[128 * 64];         // [j][d]  16 KB (d-pairs contiguous)
  __shared__ u16 lV[64 * 128];         // [d][j]  16 KB (j-pairs contiguous)
  __shared__ u16 lP[8][16 * 128];      // per-wave P tiles, 32 KB

  const int tid  = threadIdx.x;
  const int wave = tid >> 5, lane = tid & 31;
  const int mlo  = lane & 15, half = lane >> 4;
  const int b = blockIdx.z, h = blockIdx.y;
  const int i0 = blockIdx.x * 128;
  const size_t base = ((size_t)(b * HEADS + h)) * N_SP * HDIM;
  const u16* Qb = Q + base;
  const u16* Kb = K + base;
  const u16* Vb = V + base;

  // Q A-fragments: 16 rows per wave, K=64 -> 2 frags (pairs contiguous in [n][d])
  unsigned qfr[2][8];
  {
    int m = i0 + wave * 16 + mlo;
#pragma unroll
    for (int f = 0; f < 2; ++f)
#pragma unroll
      for (int v = 0; v < 8; ++v) {
        int k = f * 32 + (v >> 2) * 16 + half * 8 + (v & 3) * 2;
        qfr[f][v] = *(const unsigned*)(Qb + (size_t)m * HDIM + k);
      }
  }

  v8f oacc[4] = {};
  float mrun[8], lrun[8];
#pragma unroll
  for (int r = 0; r < 8; ++r) { mrun[r] = -1e30f; lrun[r] = 0.f; }

  for (int jt = 0; jt < 8; ++jt) {
    const int j0 = jt * 128;
    {   // stage K: 128 rows x 128B, direct copy
      int j = tid >> 1, off = (tid & 1) * 32;
      const u16* src = Kb + (size_t)(j0 + j) * HDIM + off;
      if (jt < 7) __builtin_prefetch(src + 128 * HDIM, 0, 1);
      cp128(lK + j * 64 + off,      src);
      cp128(lK + j * 64 + off + 8,  src + 8);
      cp128(lK + j * 64 + off + 16, src + 16);
      cp128(lK + j * 64 + off + 24, src + 24);
    }
    {   // stage V: 64 d-rows x 256B, direct copy
      int d = tid >> 2, off = (tid & 3) * 32;
      const u16* src = Vb + (size_t)d * N_SP + j0 + off;
      cp128(lV + d * 128 + off,      src);
      cp128(lV + d * 128 + off + 8,  src + 8);
      cp128(lV + d * 128 + off + 16, src + 16);
      cp128(lV + d * 128 + off + 24, src + 24);
    }
    async_wait();
    __syncthreads();

    // S = Q^T K : 8 j-subtiles x 2 WMMAs
    v8f s[8];
#pragma unroll
    for (int nt = 0; nt < 8; ++nt) {
      unsigned bf0[8], bf1[8];
      int n = nt * 16 + mlo;
#pragma unroll
      for (int v = 0; v < 8; ++v) {
        bf0[v] = *(const unsigned*)(lK + n * 64 + (v >> 2) * 16 + half * 8 + (v & 3) * 2);
        bf1[v] = *(const unsigned*)(lK + n * 64 + 32 + (v >> 2) * 16 + half * 8 + (v & 3) * 2);
      }
      v8f t = {};
      t = bf16_wmma(qfr[0], bf0, t);
      t = bf16_wmma(qfr[1], bf1, t);
      s[nt] = t;
    }

    // online softmax; each row lives in one 16-lane half at vgpr slot r
#pragma unroll
    for (int r = 0; r < 8; ++r) {
      float mx = -1e30f;
#pragma unroll
      for (int nt = 0; nt < 8; ++nt) mx = fmaxf(mx, s[nt][r] * SCALE_F);
      mx = half_max(mx);
      float mnew = fmaxf(mrun[r], mx);
      float corr = __expf(mrun[r] - mnew);
      float psum = 0.f;
#pragma unroll
      for (int nt = 0; nt < 8; ++nt) {
        float p = __expf(s[nt][r] * SCALE_F - mnew);
        s[nt][r] = p;
        psum += p;
      }
      psum = half_sum(psum);
      lrun[r] = lrun[r] * corr + psum;
      mrun[r] = mnew;
#pragma unroll
      for (int dt = 0; dt < 4; ++dt) oacc[dt][r] *= corr;
    }

    // P -> this wave's LDS region (per-wave region, DS in-order)
#pragma unroll
    for (int nt = 0; nt < 8; ++nt)
#pragma unroll
      for (int r = 0; r < 8; ++r)
        lP[wave][(half * 8 + r) * 128 + nt * 16 + mlo] = f2bf(s[nt][r]);

    // O += P * V^T  (4 k-frags x 4 d-subtiles)
    unsigned pfr[4][8];
#pragma unroll
    for (int f = 0; f < 4; ++f)
#pragma unroll
      for (int v = 0; v < 8; ++v) {
        int kk = f * 32 + (v >> 2) * 16 + half * 8 + (v & 3) * 2;
        pfr[f][v] = *(const unsigned*)&lP[wave][mlo * 128 + kk];
      }
#pragma unroll
    for (int dt = 0; dt < 4; ++dt) {
      int n = dt * 16 + mlo;
#pragma unroll
      for (int f = 0; f < 4; ++f) {
        unsigned bf[8];
#pragma unroll
        for (int v = 0; v < 8; ++v)
          bf[v] = *(const unsigned*)(lV + n * 128 + f * 32 + (v >> 2) * 16 + half * 8 + (v & 3) * 2);
        oacc[dt] = bf16_wmma(pfr[f], bf, oacc[dt]);
      }
    }
    __syncthreads();
  }

  // normalize and store O^T into [b][n][c]
#pragma unroll
  for (int dt = 0; dt < 4; ++dt) {
#pragma unroll
    for (int r = 0; r < 8; ++r) {
      int n = i0 + wave * 16 + half * 8 + r;
      int c = h * HDIM + dt * 16 + mlo;
      Ob[((size_t)b * N_SP + n) * C_DIM + c] = f2bf(oacc[dt][r] / lrun[r]);
    }
  }
}

// ---------------------------------------------------------------- output projection + residual
// y[b,o,n] = gamma*(sum_c Wo[o,c]*A[b,n,c] + bo[o]) + x[b,o,n]
__global__ __launch_bounds__(256) void out_gemm(const u16* __restrict__ W,
                                                const float* __restrict__ bias,
                                                const u16* __restrict__ Abuf,
                                                const float* __restrict__ x,
                                                const float* __restrict__ gamma,
                                                float* __restrict__ y) {
  __shared__ u16 lA[128 * 32];
  __shared__ u16 lB[128 * 32];

  const int tid  = threadIdx.x;
  const int wave = tid >> 5, lane = tid & 31;
  const int mlo  = lane & 15, half = lane >> 4;
  const int b  = blockIdx.z;
  const int m0 = blockIdx.y * 128;
  const int n0 = blockIdx.x * 128;
  const int wm = (wave >> 1) * 32;
  const int wn = (wave & 1) * 64;

  const u16* Ab = Abuf + (size_t)b * N_SP * C_DIM;
  v8f acc[2][4] = {};
  const float g = gamma[0];

  for (int k0 = 0; k0 < C_DIM; k0 += 32) {
    {
      int row = tid >> 1, off = (tid & 1) * 16;
      const u16* srcA = W + (size_t)(m0 + row) * C_DIM + k0 + off;
      const u16* srcB = Ab + (size_t)(n0 + row) * C_DIM + k0 + off;
      if (k0 + 32 < C_DIM) __builtin_prefetch(srcA + 32, 0, 1);
      cp128(lA + row * 32 + off,     srcA);
      cp128(lA + row * 32 + off + 8, srcA + 8);
      cp128(lB + row * 32 + off,     srcB);
      cp128(lB + row * 32 + off + 8, srcB + 8);
    }
    async_wait();
    __syncthreads();

    unsigned afr[2][8], bfr[4][8];
#pragma unroll
    for (int mi = 0; mi < 2; ++mi) {
      int m = wm + mi * 16 + mlo;
#pragma unroll
      for (int v = 0; v < 8; ++v) {
        int kk = (v >> 2) * 16 + half * 8 + (v & 3) * 2;
        afr[mi][v] = *(const unsigned*)(lA + m * 32 + kk);
      }
    }
#pragma unroll
    for (int ni = 0; ni < 4; ++ni) {
      int n = wn + ni * 16 + mlo;
#pragma unroll
      for (int v = 0; v < 8; ++v)
        bfr[ni][v] = *(const unsigned*)(lB + n * 32 + (v >> 2) * 16 + half * 8 + (v & 3) * 2);
    }
#pragma unroll
    for (int mi = 0; mi < 2; ++mi)
#pragma unroll
      for (int ni = 0; ni < 4; ++ni)
        acc[mi][ni] = bf16_wmma(afr[mi], bfr[ni], acc[mi][ni]);
    __syncthreads();
  }

#pragma unroll
  for (int mi = 0; mi < 2; ++mi) {
#pragma unroll
    for (int ni = 0; ni < 4; ++ni) {
      int n = n0 + wn + ni * 16 + mlo;
#pragma unroll
      for (int r = 0; r < 8; ++r) {
        int o = m0 + wm + mi * 16 + half * 8 + r;
        size_t idx = ((size_t)b * C_DIM + o) * N_SP + n;
        y[idx] = g * (acc[mi][ni][r] + bias[o]) + x[idx];
      }
    }
  }
}

// ---------------------------------------------------------------- launch
extern "C" void kernel_launch(void* const* d_in, const int* in_sizes, int n_in,
                              void* d_out, int out_size, void* d_ws, size_t ws_size,
                              hipStream_t stream) {
  const float* x  = (const float*)d_in[0];
  const float* wq = (const float*)d_in[1];
  const float* bq = (const float*)d_in[2];
  const float* wk = (const float*)d_in[3];
  const float* bk = (const float*)d_in[4];
  const float* wv = (const float*)d_in[5];
  const float* bv = (const float*)d_in[6];
  const float* wo = (const float*)d_in[7];
  const float* bo = (const float*)d_in[8];
  const float* gm = (const float*)d_in[9];
  float* y = (float*)d_out;

  const size_t XN = (size_t)BATCH * C_DIM * N_SP;   // 8,388,608
  const size_t WN = (size_t)C_DIM * C_DIM;          // 262,144

  u16* ws  = (u16*)d_ws;
  u16* xb  = ws;            // [b][n][c]
  u16* wqb = xb  + XN;
  u16* wkb = wqb + WN;
  u16* wvb = wkb + WN;
  u16* wob = wvb + WN;
  u16* qb  = wob + WN;      // [b][h][n][d]
  u16* kb  = qb  + XN;      // [b][h][n][d]
  u16* vb  = kb  + XN;      // [b][h][d][n]
  u16* ab  = vb  + XN;      // [b][n][c]

  cvt_x_t<<<dim3(N_SP / 32, C_DIM / 32, BATCH), 256, 0, stream>>>(x, xb);
  cvt_bf16<<<256, 256, 0, stream>>>(wq, wqb, (int)WN);
  cvt_bf16<<<256, 256, 0, stream>>>(wk, wkb, (int)WN);
  cvt_bf16<<<256, 256, 0, stream>>>(wv, wvb, (int)WN);
  cvt_bf16<<<256, 256, 0, stream>>>(wo, wob, (int)WN);

  dim3 gg(N_SP / 128, C_DIM / 128, BATCH);          // (8,4,16)
  qkv_gemm<<<gg, 256, 0, stream>>>(wqb, bq, xb, qb, 0);
  qkv_gemm<<<gg, 256, 0, stream>>>(wkb, bk, xb, kb, 0);
  qkv_gemm<<<gg, 256, 0, stream>>>(wvb, bv, xb, vb, 1);

  flash_attn<<<dim3(N_SP / 128, HEADS, BATCH), 256, 0, stream>>>(qb, kb, vb, ab);

  out_gemm<<<gg, 256, 0, stream>>>(wob, bo, ab, x, gm, y);
}
